// EncoderDecoder_65730179498212
// MI455X (gfx1250) — compile-verified, matrix operand order checked by reading, not run on hardware
//
#include <hip/hip_runtime.h>

// ---------------------------------------------------------------------------
// EncoderDecoder LSTM on gfx1250 (MI455X):
//   encoder: 168 sequential steps of  pre = x_t @ W_ih^T + h @ W_hh^T + b
//   decoder: 24 sequential steps of   pre = h @ W_hh_d^T + b
//   head:    per-step dense (OUT=64) on decoder h's
// Strategy: bf16 WMMA (v_wmma_f32_16x16x32_bf16) with fp32 accum, one kernel
// launch per step (graph-friendly). All state (~32 MB) is L2-resident, so this
// is a latency problem: stream B (weights) from L2 with b128 loads + WGP-scope
// prefetch, stage the shared A tile (h) in LDS via async global->LDS copies.
// ---------------------------------------------------------------------------

typedef __attribute__((ext_vector_type(16))) __bf16 bf16x16;
typedef __attribute__((ext_vector_type(8)))  __bf16 bf16x8;
typedef __attribute__((ext_vector_type(8)))  float  f32x8;

// Type used by the async global->LDS builtin (per hipcc diagnostic):
// first param is v4i addrspace(1)*, second v4i addrspace(3)*.
typedef int v4i_t __attribute__((vector_size(16)));
typedef __attribute__((address_space(1))) v4i_t* v4i_gptr;
typedef __attribute__((address_space(3))) v4i_t* v4i_lptr;

#if __has_builtin(__builtin_amdgcn_global_load_async_to_lds_b128) && \
    __has_builtin(__builtin_amdgcn_s_wait_asynccnt)
#define USE_ASYNC_LDS 1
#else
#define USE_ASYNC_LDS 0
#endif

namespace {
constexpr int NB    = 512;   // batch
constexpr int CIN   = 64;    // in channels
constexpr int TT    = 168;   // encode length
constexpr int HH    = 512;   // hidden
constexpr int NSTEP = 24;    // decoder steps
constexpr int OUTC  = 64;    // out channels

// LDS A-tile stride: 520 elems = 1040 B = 260 DWORDs; 260 % 64 = 4, so the 16
// lanes of a half-wave land on banks 0,4,...,60 and each ds_load_b128 spans 4
// consecutive banks -> conflict-free.
constexpr int ASTR  = HH + 8;

// workspace layout (bytes)
constexpr size_t OFF_XT   = 0;                                   // (T,N,C) bf16
constexpr size_t SZ_XT    = (size_t)TT * NB * CIN * 2;
constexpr size_t OFF_WIH  = OFF_XT + SZ_XT;                      // (4H,C) bf16
constexpr size_t SZ_WIH   = (size_t)4 * HH * CIN * 2;
constexpr size_t OFF_WHE  = OFF_WIH + SZ_WIH;                    // (4H,H) bf16
constexpr size_t SZ_WHH   = (size_t)4 * HH * HH * 2;
constexpr size_t OFF_WHD  = OFF_WHE + SZ_WHH;                    // (4H,H) bf16
constexpr size_t OFF_WD   = OFF_WHD + SZ_WHH;                    // (S,OUT,H) bf16
constexpr size_t SZ_WD    = (size_t)NSTEP * OUTC * HH * 2;
constexpr size_t OFF_H0   = OFF_WD + SZ_WD;                      // (N,H) bf16 ping
constexpr size_t SZ_H     = (size_t)NB * HH * 2;
constexpr size_t OFF_H1   = OFF_H0 + SZ_H;                       // (N,H) bf16 pong
constexpr size_t OFF_C    = OFF_H1 + SZ_H;                       // (N,H) f32
constexpr size_t SZ_C     = (size_t)NB * HH * 4;
constexpr size_t OFF_HS   = OFF_C + SZ_C;                        // (S,N,H) bf16
constexpr size_t SZ_HS    = (size_t)NSTEP * NB * HH * 2;
constexpr size_t WS_NEED  = OFF_HS + SZ_HS;
}

// ---- WMMA fragment loaders ------------------------------------------------
// 16x32 bf16 A-fragment (B mirrored for columns): lane L<16 holds row/col
// (base + L), K chunks [k0,k0+8) and [k0+16,k0+24); lane L+16 holds the same
// row/col with K chunks [k0+8,k0+16) and [k0+24,k0+32). Two contiguous
// 16-byte loads each in row-major storage.
__device__ __forceinline__ bf16x16 load_frag(const __bf16* __restrict__ base,
                                             int ld, int row_base, int k0, int lane) {
  const int r  = row_base + (lane & 15);
  const int kb = k0 + ((lane >> 4) << 3);
  const __bf16* p = base + (size_t)r * ld + kb;
  bf16x8 lo = *(const bf16x8*)(p);
  bf16x8 hi = *(const bf16x8*)(p + 16);
  return __builtin_shufflevector(lo, hi, 0,1,2,3,4,5,6,7,8,9,10,11,12,13,14,15);
}

// Same fragment, from the padded LDS A tile (row_base is tile-local).
__device__ __forceinline__ bf16x16 load_frag_lds(const __bf16* base,
                                                 int row_base, int k0, int lane) {
  const int r  = row_base + (lane & 15);
  const int kb = k0 + ((lane >> 4) << 3);
  const __bf16* p = base + r * ASTR + kb;
  bf16x8 lo = *(const bf16x8*)(p);
  bf16x8 hi = *(const bf16x8*)(p + 16);
  return __builtin_shufflevector(lo, hi, 0,1,2,3,4,5,6,7,8,9,10,11,12,13,14,15);
}

__device__ __forceinline__ float sigm_f(float x) {
  return 1.0f / (1.0f + __expf(-x));
}
__device__ __forceinline__ float tanh_f(float x) {
  // tanh(x) = 1 - 2/(exp(2x)+1); saturates correctly at +/-1
  return 1.0f - 2.0f / (__expf(2.0f * x) + 1.0f);
}

// ---- one LSTM step: pre = h_in @ Whh^T (+ x_t @ Wih^T) + b; gate update ----
// grid: 64 WGs (8 batch-blocks x 8 hidden-blocks of 64), block: 256 (8 waves)
// wave w: row-block rb = w&3 (16 rows), col-blocks cb0,cb0+1 (2x16 cols),
// 4 gate accumulators per output tile.
__global__ __launch_bounds__(256)
void lstm_step_kernel(const __bf16* __restrict__ h_in,
                      __bf16* __restrict__ h_out,
                      float* __restrict__ c_io,
                      const __bf16* __restrict__ Whh,
                      const float* __restrict__ bias,
                      const __bf16* __restrict__ x_t,
                      const __bf16* __restrict__ Wih,
                      int has_x, int c_zero) {
  __shared__ __align__(16) __bf16 Ash[64 * ASTR];   // 65 KB, fine on gfx1250

  const int tid  = threadIdx.x;
  const int lane = tid & 31;
  const int w    = tid >> 5;
  const int wgM  = (blockIdx.x >> 3) * 64;   // batch-row base
  const int wgN  = (blockIdx.x & 7) * 64;    // hidden-col base
  const int rb   = w & 3;
  const int cb0  = (w >> 2) << 1;

  // ---- stage the 64x512 A tile (h rows for this WG) into LDS --------------
  const __bf16* srcA = h_in + (size_t)wgM * HH;
#if USE_ASYNC_LDS
  for (int i = tid; i < 64 * (HH / 8); i += 256) {          // 4096 b128 chunks
    const int row = i >> 6;
    const int ck  = (i & 63) << 3;                          // elem offset in row
    __builtin_amdgcn_global_load_async_to_lds_b128(
        (v4i_gptr)(srcA + (size_t)row * HH + ck),
        (v4i_lptr)(Ash + row * ASTR + ck),
        0, 0);
  }
  __builtin_amdgcn_s_wait_asynccnt(0);
#else
  for (int i = tid; i < 64 * (HH / 8); i += 256) {
    const int row = i >> 6;
    const int ck  = (i & 63) << 3;
    *(bf16x8*)(Ash + row * ASTR + ck) = *(const bf16x8*)(srcA + (size_t)row * HH + ck);
  }
#endif
  __syncthreads();

  f32x8 acc[2][4];
  f32x8 z = {};
#pragma unroll
  for (int cb = 0; cb < 2; ++cb)
#pragma unroll
    for (int g = 0; g < 4; ++g) acc[cb][g] = z;

  const int rowA = wgM + rb * 16;          // global batch row of this wave
  const int rowL = rb * 16;                // LDS-local row

  // recurrent GEMM: K = HH (A from LDS, B streamed from L2)
  for (int k0 = 0; k0 < HH; k0 += 32) {
    bf16x16 a = load_frag_lds(Ash, rowL, k0, lane);
    if (k0 + 32 < HH) {
      // WGP-scope prefetch of the next K-chunk of weights (locality 3)
      __builtin_prefetch(Whh + (size_t)(wgN + (lane & 15)) * HH + k0 + 32, 0, 3);
    }
#pragma unroll
    for (int cb = 0; cb < 2; ++cb) {
#pragma unroll
      for (int g = 0; g < 4; ++g) {
        const int colG = g * HH + wgN + (cb0 + cb) * 16;   // B column = gate row of Whh
        bf16x16 b = load_frag(Whh, HH, colG, k0, lane);
        acc[cb][g] = __builtin_amdgcn_wmma_f32_16x16x32_bf16(
            false, a, false, b, (short)0, acc[cb][g], false, false);
      }
    }
  }

  // input GEMM: K = CIN (encoder only; x read once, direct loads)
  if (has_x) {
    for (int k0 = 0; k0 < CIN; k0 += 32) {
      bf16x16 a = load_frag(x_t, CIN, rowA, k0, lane);
#pragma unroll
      for (int cb = 0; cb < 2; ++cb) {
#pragma unroll
        for (int g = 0; g < 4; ++g) {
          const int colG = g * HH + wgN + (cb0 + cb) * 16;
          bf16x16 b = load_frag(Wih, CIN, colG, k0, lane);
          acc[cb][g] = __builtin_amdgcn_wmma_f32_16x16x32_bf16(
              false, a, false, b, (short)0, acc[cb][g], false, false);
        }
      }
    }
  }

  // gate math + state update.  D-tile layout: acc[j] lane l -> row j+(l>>4)*8,
  // col = l&15 within the 16x16 tile.
#pragma unroll
  for (int cb = 0; cb < 2; ++cb) {
    const int col = wgN + (cb0 + cb) * 16 + (lane & 15);
    const float bi  = bias[0 * HH + col];
    const float bf_ = bias[1 * HH + col];
    const float bg  = bias[2 * HH + col];
    const float bo  = bias[3 * HH + col];
#pragma unroll
    for (int j = 0; j < 8; ++j) {
      const int row = rowA + j + ((lane >> 4) << 3);
      const size_t idx = (size_t)row * HH + col;
      const float cp = c_zero ? 0.0f : c_io[idx];
      const float gi = sigm_f(acc[cb][0][j] + bi);
      const float gf = sigm_f(acc[cb][1][j] + bf_);
      const float gg = tanh_f(acc[cb][2][j] + bg);
      const float go = sigm_f(acc[cb][3][j] + bo);
      const float cn = gf * cp + gi * gg;
      c_io[idx] = cn;
      h_out[idx] = (__bf16)(go * tanh_f(cn));
    }
  }
}

// ---- dense head: out[n,o,t] = hs[t,n,:] . Wd[t,o,:] + bd[t,o] --------------
// grid: NSTEP * 8 WGs (64 batch rows each), OUT=64 covered by one WG.
__global__ __launch_bounds__(256)
void dense_kernel(const __bf16* __restrict__ hs,
                  const __bf16* __restrict__ Wd,
                  const float* __restrict__ bd,
                  float* __restrict__ out) {
  const int lane = threadIdx.x & 31;
  const int w    = threadIdx.x >> 5;
  const int t    = blockIdx.x >> 3;
  const int wgM  = (blockIdx.x & 7) * 64;
  const int rb   = w & 3;
  const int cb0  = (w >> 2) << 1;
  const int rowA = wgM + rb * 16;

  const __bf16* A = hs + (size_t)t * NB * HH;
  const __bf16* B = Wd + (size_t)t * OUTC * HH;

  f32x8 acc[2];
  f32x8 z = {};
  acc[0] = z; acc[1] = z;

  for (int k0 = 0; k0 < HH; k0 += 32) {
    bf16x16 a = load_frag(A, HH, rowA, k0, lane);
#pragma unroll
    for (int cb = 0; cb < 2; ++cb) {
      bf16x16 b = load_frag(B, HH, (cb0 + cb) * 16, k0, lane);
      acc[cb] = __builtin_amdgcn_wmma_f32_16x16x32_bf16(
          false, a, false, b, (short)0, acc[cb], false, false);
    }
  }

#pragma unroll
  for (int cb = 0; cb < 2; ++cb) {
    const int o = (cb0 + cb) * 16 + (lane & 15);
    const float bb = bd[t * OUTC + o];
#pragma unroll
    for (int j = 0; j < 8; ++j) {
      const int row = rowA + j + ((lane >> 4) << 3);
      out[(size_t)row * OUTC * NSTEP + (size_t)o * NSTEP + t] = acc[cb][j] + bb;
    }
  }
}

// ---- prep kernels ----------------------------------------------------------
__global__ void cvt_bf16_kernel(const float* __restrict__ s, __bf16* __restrict__ d, int n) {
  for (int i = blockIdx.x * blockDim.x + threadIdx.x; i < n; i += gridDim.x * blockDim.x)
    d[i] = (__bf16)s[i];
}

// xT[t][n][c] = x[n][c][t]
__global__ void transpose_x_kernel(const float* __restrict__ x, __bf16* __restrict__ xT) {
  const int total = TT * NB * CIN;
  for (int i = blockIdx.x * blockDim.x + threadIdx.x; i < total; i += gridDim.x * blockDim.x) {
    const int c = i % CIN;
    const int n = (i / CIN) % NB;
    const int t = i / (CIN * NB);
    xT[i] = (__bf16)x[(size_t)n * CIN * TT + (size_t)c * TT + t];
  }
}

__global__ void zero_bf16_kernel(__bf16* __restrict__ p, int n) {
  for (int i = blockIdx.x * blockDim.x + threadIdx.x; i < n; i += gridDim.x * blockDim.x)
    p[i] = (__bf16)0.0f;
}

// ---------------------------------------------------------------------------
extern "C" void kernel_launch(void* const* d_in, const int* in_sizes, int n_in,
                              void* d_out, int out_size, void* d_ws, size_t ws_size,
                              hipStream_t stream) {
  if (ws_size < WS_NEED) return;

  const float* x     = (const float*)d_in[0];
  const float* WihF  = (const float*)d_in[1];
  const float* WheF  = (const float*)d_in[2];
  const float* b_e   = (const float*)d_in[3];
  const float* WhdF  = (const float*)d_in[4];
  const float* b_d   = (const float*)d_in[5];
  const float* WdF   = (const float*)d_in[6];
  const float* b_dns = (const float*)d_in[7];
  float* out = (float*)d_out;

  char* ws = (char*)d_ws;
  __bf16* xT  = (__bf16*)(ws + OFF_XT);
  __bf16* Wih = (__bf16*)(ws + OFF_WIH);
  __bf16* Whe = (__bf16*)(ws + OFF_WHE);
  __bf16* Whd = (__bf16*)(ws + OFF_WHD);
  __bf16* Wd  = (__bf16*)(ws + OFF_WD);
  __bf16* h0  = (__bf16*)(ws + OFF_H0);
  __bf16* h1  = (__bf16*)(ws + OFF_H1);
  float*  cB  = (float*)(ws + OFF_C);
  __bf16* hs  = (__bf16*)(ws + OFF_HS);
  __bf16* hbuf[2] = { h0, h1 };

  // prep: bf16 conversions + x transpose + h0 = 0
  cvt_bf16_kernel<<<256, 256, 0, stream>>>(WihF, Wih, 4 * HH * CIN);
  cvt_bf16_kernel<<<512, 256, 0, stream>>>(WheF, Whe, 4 * HH * HH);
  cvt_bf16_kernel<<<512, 256, 0, stream>>>(WhdF, Whd, 4 * HH * HH);
  cvt_bf16_kernel<<<512, 256, 0, stream>>>(WdF,  Wd,  NSTEP * OUTC * HH);
  transpose_x_kernel<<<1024, 256, 0, stream>>>(x, xT);
  zero_bf16_kernel<<<256, 256, 0, stream>>>(h0, NB * HH);

  // encoder: 168 sequential steps (h ping-pong, c in place; c starts 0)
  for (int t = 0; t < TT; ++t) {
    lstm_step_kernel<<<64, 256, 0, stream>>>(
        hbuf[t & 1], hbuf[(t + 1) & 1], cB, Whe, b_e,
        xT + (size_t)t * NB * CIN, Wih, /*has_x=*/1, /*c_zero=*/(t == 0));
  }
  // after 168 steps, h_enc is in hbuf[0]

  // decoder: 24 sequential steps, c restarts at 0, outputs recorded in hs
  for (int s = 0; s < NSTEP; ++s) {
    const __bf16* hin = (s == 0) ? hbuf[0] : (hs + (size_t)(s - 1) * NB * HH);
    __bf16* hout = hs + (size_t)s * NB * HH;
    lstm_step_kernel<<<64, 256, 0, stream>>>(
        hin, hout, cB, Whd, b_d, nullptr, nullptr, /*has_x=*/0, /*c_zero=*/(s == 0));
  }

  // per-step dense head
  dense_kernel<<<NSTEP * 8, 256, 0, stream>>>(hs, Wd, b_dns, out);
}
